// HorizonLSTMWithVariance_64072322122513
// MI455X (gfx1250) — compile-verified, matrix operand order checked by reading, not run on hardware
//
#include <hip/hip_runtime.h>
#include <cmath>

// Problem constants
#define B_   64
#define T_   256
#define IN_  13
#define H_   512
#define HH_  256
#define L_   4
#define OUT_ 3
// 4*H gate width
#define G_   2048
// elements per (layer) h/c plane
#define PLANE_ (B_ * H_)          // 64*512 = 32768
#define TSTRIDE_ ((long)L_ * PLANE_) // per-timestep stride in Hhist (halves)

typedef _Float16 h16v __attribute__((ext_vector_type(16)));
typedef _Float16 h8v  __attribute__((ext_vector_type(8)));
typedef float    f8v  __attribute__((ext_vector_type(8)));

// ---------------------------------------------------------------------------
// CDNA5 WMMA 16x16x32 f16 operand loader.
// A operand (16xK=32, M x K, row-major, stride ld):
//   lane L: row = row0 + (L&15), koff = (L>>4)*8
//   elems [0..7]  = A[row][k0+koff .. +7]
//   elems [8..15] = A[row][k0+16+koff .. +7]
// B operand (W stored N x K row-major) uses the identical mapping with
// row -> output column n (B[k][n] = W[n][k]).
// ---------------------------------------------------------------------------
__device__ __forceinline__ h16v load_tile16x32(const _Float16* __restrict__ base,
                                               int row0, int k0, int ld) {
  const int lane = threadIdx.x & 31;
  const int r    = row0 + (lane & 15);
  const int koff = (lane >> 4) << 3; // 0 or 8
  const _Float16* p = base + (long)r * ld + k0 + koff;
  h8v lo = *reinterpret_cast<const h8v*>(p);
  h8v hi = *reinterpret_cast<const h8v*>(p + 16);
  h16v v;
#pragma unroll
  for (int i = 0; i < 8; ++i) { v[i] = lo[i]; v[i + 8] = hi[i]; }
  return v;
}

// Blocked-A loader for the heads over Hhist[t][l][m][h]:
// logical row = t*64 + m, logical col k = l*512 + h
// addr = base + (row>>6)*rbs + (row&63)*512 + (k>>9)*cbs + (k&511)
// K tiles (32-aligned) never straddle a 512 block since 512 % 32 == 0.
__device__ __forceinline__ h16v load_tileA_blocked(const _Float16* __restrict__ base,
                                                   int row0, int k0,
                                                   long rbs, long cbs) {
  const int lane = threadIdx.x & 31;
  const int r    = row0 + (lane & 15);
  const int k    = k0 + ((lane >> 4) << 3);
  const _Float16* p = base + (long)(r >> 6) * rbs + (long)(r & 63) * 512
                           + (long)(k >> 9) * cbs + (k & 511);
  h8v lo = *reinterpret_cast<const h8v*>(p);
  h8v hi = *reinterpret_cast<const h8v*>(p + 16);
  h16v v;
#pragma unroll
  for (int i = 0; i < 8; ++i) { v[i] = lo[i]; v[i + 8] = hi[i]; }
  return v;
}

__device__ __forceinline__ float sigmoidf_(float x) {
  return 1.0f / (1.0f + __expf(-x));
}

// ---------------------------------------------------------------------------
// Prep: fp32->fp16 weight conversion, bias fold, state zeroing.
// ---------------------------------------------------------------------------
__global__ void prep_kernel(const float* __restrict__ Wih, const float* __restrict__ Whh,
                            const float* __restrict__ Wout1, const float* __restrict__ Wv1,
                            const float* __restrict__ bih, const float* __restrict__ bhh,
                            _Float16* __restrict__ Wih16, _Float16* __restrict__ Whh16,
                            _Float16* __restrict__ Wout116, _Float16* __restrict__ Wv116,
                            float* __restrict__ bias, float* __restrict__ Cst,
                            _Float16* __restrict__ Hzero) {
  const long stride = (long)gridDim.x * blockDim.x;
  const long t0 = (long)blockIdx.x * blockDim.x + threadIdx.x;
  const long NW = (long)L_ * G_ * H_; // 4.19M per matrix
  for (long i = t0; i < NW; i += stride) {
    Wih16[i] = (_Float16)Wih[i];
    Whh16[i] = (_Float16)Whh[i];
  }
  for (long i = t0; i < (long)HH_ * H_; i += stride) Wout116[i] = (_Float16)Wout1[i];
  for (long i = t0; i < (long)HH_ * (L_ * H_); i += stride) Wv116[i] = (_Float16)Wv1[i];
  for (long i = t0; i < (long)L_ * G_; i += stride) bias[i] = bih[i] + bhh[i];
  for (long i = t0; i < (long)L_ * PLANE_; i += stride) {
    Cst[i] = 0.0f;
    Hzero[i] = (_Float16)0.0f;
  }
}

// ---------------------------------------------------------------------------
// Input MLP on x[:,0,:] only (that is all the reference consumes).
// ---------------------------------------------------------------------------
__global__ void mlp1_kernel(const float* __restrict__ x, const float* __restrict__ W1,
                            const float* __restrict__ b1, float* __restrict__ t1) {
  const int m = blockIdx.x;      // 64
  const int j = threadIdx.x;     // 256
  const float* xr = x + (long)m * T_ * IN_;  // x[m][0][:]
  const float* w  = W1 + j * IN_;
  float s = b1[j];
#pragma unroll
  for (int k = 0; k < IN_; ++k) s += xr[k] * w[k];
  t1[m * HH_ + j] = s > 0.0f ? s : 0.0f;
}

__global__ void mlp2_kernel(const float* __restrict__ t1, const float* __restrict__ W2,
                            const float* __restrict__ b2, _Float16* __restrict__ x0) {
  const int idx = blockIdx.x * blockDim.x + threadIdx.x; // 64*512
  const int m = idx >> 9, j = idx & 511;
  const float* tr = t1 + m * HH_;
  const float* w  = W2 + (long)j * HH_;
  float s = b2[j];
  for (int k = 0; k < HH_; ++k) s += tr[k] * w[k];
  x0[idx] = (_Float16)s;
}

// ---------------------------------------------------------------------------
// Fused LSTM step for one (t, layer): G = Ain*Wih^T + Hprev*Whh^T, then the
// cell nonlinearity. Each workgroup owns a 16-wide hidden slice; its 8 waves
// cover {4 gates} x {2 M-halves}; each wave computes 2 (16x16) tiles over
// K=512 with v_wmma_f32_16x16x32_f16. Gates staged through LDS, elementwise
// epilogue updates c (f32) and writes h (f16) for the next matmuls + heads.
// Grid: 32 blocks x 256 threads.
// ---------------------------------------------------------------------------
__global__ void lstm_step_kernel(const _Float16* __restrict__ Ain,    // 64x512
                                 const _Float16* __restrict__ Hprev,  // 64x512
                                 const _Float16* __restrict__ Wih,    // 2048x512
                                 const _Float16* __restrict__ Whh,    // 2048x512
                                 const float*    __restrict__ bias,   // 2048
                                 float*          __restrict__ Cst,    // 64x512
                                 _Float16*       __restrict__ Hout) { // 64x512
  __shared__ float lds_g[4][64][17]; // padded to dodge bank conflicts

  const int tid   = threadIdx.x;
  const int wave  = tid >> 5;
  const int lane  = tid & 31;
  const int hbase = blockIdx.x * 16;     // hidden slice of 16 units
  const int gate  = wave >> 1;           // 0..3
  const int msel  = wave & 1;            // which pair of M tiles
  const int n0    = gate * H_ + hbase;   // weight-row / gate-column base

  f8v acc[2];
  acc[0] = (f8v){};
  acc[1] = (f8v){};

  for (int k0 = 0; k0 < H_; k0 += 32) {
    if (k0 + 32 < H_) {
      __builtin_prefetch(Wih + (long)n0 * H_ + k0 + 32, 0, 0);
      __builtin_prefetch(Whh + (long)n0 * H_ + k0 + 32, 0, 0);
    }
    h16v b_ih = load_tile16x32(Wih, n0, k0, H_);
    h16v b_hh = load_tile16x32(Whh, n0, k0, H_);
#pragma unroll
    for (int i = 0; i < 2; ++i) {
      const int m0 = (msel * 2 + i) * 16;
      h16v a1 = load_tile16x32(Ain, m0, k0, H_);
      acc[i] = __builtin_amdgcn_wmma_f32_16x16x32_f16(false, a1, false, b_ih,
                                                      (short)0, acc[i], false, false);
      h16v a2 = load_tile16x32(Hprev, m0, k0, H_);
      acc[i] = __builtin_amdgcn_wmma_f32_16x16x32_f16(false, a2, false, b_hh,
                                                      (short)0, acc[i], false, false);
    }
  }

  // C/D layout: lane L, vgpr r -> M = tile_m0 + r + 8*(L>>4), N = n0 + (L&15)
  const int jl    = lane & 15;
  const int mhalf = (lane >> 4) * 8;
#pragma unroll
  for (int i = 0; i < 2; ++i) {
    const int m0 = (msel * 2 + i) * 16;
#pragma unroll
    for (int r = 0; r < 8; ++r)
      lds_g[gate][m0 + mhalf + r][jl] = acc[i][r];
  }
  __syncthreads();

  for (int idx = tid; idx < 64 * 16; idx += 256) {
    const int m = idx >> 4, jloc = idx & 15;
    const int j = hbase + jloc;
    const float gi = lds_g[0][m][jloc] + bias[j];
    const float gf = lds_g[1][m][jloc] + bias[H_ + j];
    const float gg = lds_g[2][m][jloc] + bias[2 * H_ + j];
    const float go = lds_g[3][m][jloc] + bias[3 * H_ + j];
    const float cprev = Cst[m * H_ + j];
    const float cn = sigmoidf_(gf) * cprev + sigmoidf_(gi) * tanhf(gg);
    const float hn = sigmoidf_(go) * tanhf(cn);
    Cst[m * H_ + j]  = cn;
    Hout[m * H_ + j] = (_Float16)hn;
  }
}

// ---------------------------------------------------------------------------
// Head GEMM: D = relu(A @ W^T + b), A is (16384 x K) viewed over Hhist with
// blocked addressing, W is (256 x K) f16, D stored f16 (16384 x 256).
// Block: 64 rows x 128 cols (8 waves x {1 ntile, 4 mtiles}).
// Grid: (16384/64, 256/128) = (256, 2).
// ---------------------------------------------------------------------------
__global__ void head_gemm_kernel(const _Float16* __restrict__ Abase, long rbs, long cbs,
                                 int K, const _Float16* __restrict__ W,
                                 const float* __restrict__ bvec,
                                 _Float16* __restrict__ Dout) {
  const int tid  = threadIdx.x;
  const int wave = tid >> 5;
  const int lane = tid & 31;
  const int m0b  = blockIdx.x * 64;
  const int n0   = blockIdx.y * 128 + wave * 16;

  f8v acc[4];
#pragma unroll
  for (int mt = 0; mt < 4; ++mt) acc[mt] = (f8v){};

  for (int k0 = 0; k0 < K; k0 += 32) {
    h16v b = load_tile16x32(W, n0, k0, K);
#pragma unroll
    for (int mt = 0; mt < 4; ++mt) {
      h16v a = load_tileA_blocked(Abase, m0b + mt * 16, k0, rbs, cbs);
      acc[mt] = __builtin_amdgcn_wmma_f32_16x16x32_f16(false, a, false, b,
                                                       (short)0, acc[mt], false, false);
    }
  }

  const int ncol = n0 + (lane & 15);
  const float bv = bvec[ncol];
#pragma unroll
  for (int mt = 0; mt < 4; ++mt) {
#pragma unroll
    for (int r = 0; r < 8; ++r) {
      const int row = m0b + mt * 16 + (lane >> 4) * 8 + r;
      float v = acc[mt][r] + bv;
      v = v > 0.0f ? v : 0.0f;
      Dout[(long)row * HH_ + ncol] = (_Float16)v;
    }
  }
}

// ---------------------------------------------------------------------------
// Final N=3 projection (tiny): out[m][t][col_off + c] from A (16384 x 256).
// ---------------------------------------------------------------------------
__global__ void head_out_kernel(const _Float16* __restrict__ A,
                                const float* __restrict__ W2,  // 3 x 256
                                const float* __restrict__ b2,  // 3
                                float* __restrict__ out,       // (B, T, 6)
                                int do_abs, int col_off) {
  const int idx = blockIdx.x * blockDim.x + threadIdx.x; // 16384*3
  if (idx >= B_ * T_ * OUT_) return;
  const int row = idx / OUT_;
  const int c   = idx - row * OUT_;
  const _Float16* a = A + (long)row * HH_;
  const float*    w = W2 + c * HH_;
  float s = 0.0f;
  for (int k = 0; k < HH_; ++k) s += (float)a[k] * w[k];
  s += b2[c];
  if (do_abs) s = fabsf(s);
  const int t = row >> 6, m = row & 63;
  out[((long)m * T_ + t) * 6 + col_off + c] = s;
}

// ---------------------------------------------------------------------------
extern "C" void kernel_launch(void* const* d_in, const int* in_sizes, int n_in,
                              void* d_out, int out_size, void* d_ws, size_t ws_size,
                              hipStream_t stream) {
  const float* x      = (const float*)d_in[0];
  const float* W_in1  = (const float*)d_in[1];
  const float* b_in1  = (const float*)d_in[2];
  const float* W_in2  = (const float*)d_in[3];
  const float* b_in2  = (const float*)d_in[4];
  const float* Wih    = (const float*)d_in[5];
  const float* Whh    = (const float*)d_in[6];
  const float* bih    = (const float*)d_in[7];
  const float* bhh    = (const float*)d_in[8];
  const float* W_out1 = (const float*)d_in[9];
  const float* b_out1 = (const float*)d_in[10];
  const float* W_out2 = (const float*)d_in[11];
  const float* b_out2 = (const float*)d_in[12];
  const float* Wv1    = (const float*)d_in[13];
  const float* bv1    = (const float*)d_in[14];
  const float* Wv2    = (const float*)d_in[15];
  const float* bv2    = (const float*)d_in[16];
  (void)in_sizes; (void)n_in; (void)out_size;

  char* ws = (char*)d_ws;
  size_t off = 0;
  auto take = [&](size_t bytes) -> char* {
    char* p = ws + off;
    off += (bytes + 255) & ~(size_t)255;
    return p;
  };
  _Float16* Wih16   = (_Float16*)take((size_t)L_ * G_ * H_ * 2);       // 8.39 MB
  _Float16* Whh16   = (_Float16*)take((size_t)L_ * G_ * H_ * 2);       // 8.39 MB
  _Float16* Wout116 = (_Float16*)take((size_t)HH_ * H_ * 2);           // 256 KB
  _Float16* Wv116   = (_Float16*)take((size_t)HH_ * L_ * H_ * 2);      // 1 MB
  float*    bias    = (float*)   take((size_t)L_ * G_ * 4);            // 32 KB
  _Float16* x0f16   = (_Float16*)take((size_t)PLANE_ * 2);             // 64 KB
  float*    t1buf   = (float*)   take((size_t)B_ * HH_ * 4);           // 64 KB
  float*    Cst     = (float*)   take((size_t)L_ * PLANE_ * 4);        // 512 KB
  _Float16* Hzero   = (_Float16*)take((size_t)L_ * PLANE_ * 2);        // 256 KB
  _Float16* Hhist   = (_Float16*)take((size_t)T_ * L_ * PLANE_ * 2);   // 67 MB
  _Float16* A1M     = (_Float16*)take((size_t)B_ * T_ * HH_ * 2);      // 8.39 MB
  _Float16* A1V     = (_Float16*)take((size_t)B_ * T_ * HH_ * 2);      // 8.39 MB
  if (off > ws_size) return; // insufficient workspace: nothing safe to do

  // 1) Weight conversion / bias fold / state zeroing (every call: graph-safe).
  prep_kernel<<<2048, 256, 0, stream>>>(Wih, Whh, W_out1, Wv1, bih, bhh,
                                        Wih16, Whh16, Wout116, Wv116,
                                        bias, Cst, Hzero);
  // 2) Input MLP -> x0 (fp16), the constant LSTM input.
  mlp1_kernel<<<B_, HH_, 0, stream>>>(x, W_in1, b_in1, t1buf);
  mlp2_kernel<<<(B_ * H_) / 256, 256, 0, stream>>>(t1buf, W_in2, b_in2, x0f16);

  // 3) The recurrent scan: one fused WMMA kernel per (t, layer).
  for (int t = 0; t < T_; ++t) {
    for (int l = 0; l < L_; ++l) {
      const _Float16* Ain = (l == 0) ? x0f16
                                     : Hhist + ((long)t * L_ + (l - 1)) * PLANE_;
      const _Float16* Hp  = (t == 0) ? Hzero + (long)l * PLANE_
                                     : Hhist + ((long)(t - 1) * L_ + l) * PLANE_;
      lstm_step_kernel<<<32, 256, 0, stream>>>(
          Ain, Hp,
          Wih16 + (long)l * G_ * H_, Whh16 + (long)l * G_ * H_,
          bias + l * G_, Cst + (long)l * PLANE_,
          Hhist + ((long)t * L_ + l) * PLANE_);
    }
  }

  // 4) Heads. Mean head reads Hhist[t][3][m][:], var head reads all layers
  //    (ca[m][l*H+h]) via blocked addressing.
  head_gemm_kernel<<<dim3(B_ * T_ / 64, 2), 256, 0, stream>>>(
      Hhist + 3L * PLANE_, TSTRIDE_, 0L, H_, Wout116, b_out1, A1M);
  head_gemm_kernel<<<dim3(B_ * T_ / 64, 2), 256, 0, stream>>>(
      Hhist, TSTRIDE_, (long)PLANE_, L_ * H_, Wv116, bv1, A1V);

  const int nfin = B_ * T_ * OUT_;
  head_out_kernel<<<(nfin + 255) / 256, 256, 0, stream>>>(
      A1M, W_out2, b_out2, (float*)d_out, 0, 0);
  head_out_kernel<<<(nfin + 255) / 256, 256, 0, stream>>>(
      A1V, Wv2, bv2, (float*)d_out, 1, OUT_);
}
